// PointPillarDiscoNet_29145648070956
// MI455X (gfx1250) — matrix operand order repaired and checked by reading, not compiled, not run on hardware
//
#include <hip/hip_runtime.h>
#include <hip/hip_bf16.h>

// ---------------- problem constants ----------------
#define BB 4
#define NN 5
#define CC 384
#define HH 96
#define WW 192
#define HWSZ (HH * WW)          // 18432
#define H0F 192.0f
#define W0F 384.0f
#define DRATIO 0.4f
#define EPSV 1e-5f

#define K1 768                  // 2*C input channels of conv1
#define M1 128
#define M2 32
#define M3 8
#define PIXB 128                // pixels per block (8 WMMA n-tiles)
#define NT 8                    // n-tiles per block

typedef __attribute__((ext_vector_type(16))) __bf16 v16bf;
typedef __attribute__((ext_vector_type(8)))  float  v8f;

union FragBF { v16bf v; unsigned int u[8]; };

__device__ __forceinline__ v8f zero8() {
  v8f z;
#pragma unroll
  for (int i = 0; i < 8; ++i) z[i] = 0.0f;
  return z;
}

// round-to-nearest-even f32 -> bf16 (as u16)
__device__ __forceinline__ unsigned short f2bf(float f) {
  unsigned u = __float_as_uint(f);
  u += 0x7fffu + ((u >> 16) & 1u);
  return (unsigned short)(u >> 16);
}
// pack two f32 into one dword of two bf16 (lo = a, hi = b)
__device__ __forceinline__ unsigned pack_bf16(float a, float b) {
  unsigned ua = __float_as_uint(a); ua += 0x7fffu + ((ua >> 16) & 1u);
  unsigned ub = __float_as_uint(b); ub += 0x7fffu + ((ub >> 16) & 1u);
  return (ua >> 16) | (ub & 0xffff0000u);
}
__device__ __forceinline__ float bflo(unsigned u) { return __uint_as_float(u << 16); }
__device__ __forceinline__ float bfhi(unsigned u) { return __uint_as_float(u & 0xffff0000u); }
// interleave low/high u16s of two dwords: {lo16(a), lo16(b)} / {hi16(a), hi16(b)}
__device__ __forceinline__ unsigned ilv_lo(unsigned a, unsigned b) {
  return (a & 0xffffu) | (b << 16);
}
__device__ __forceinline__ unsigned ilv_hi(unsigned a, unsigned b) {
  return (a >> 16) | (b & 0xffff0000u);
}

// ---------------------------------------------------------------
// Prep: convert weights to bf16, fold conv-bias + BN into scale/shift
// ---------------------------------------------------------------
__global__ __launch_bounds__(256) void prep_kernel(
    const float* __restrict__ w1, const float* __restrict__ b1,
    const float* __restrict__ g1, const float* __restrict__ be1,
    const float* __restrict__ m1, const float* __restrict__ v1,
    const float* __restrict__ w2, const float* __restrict__ b2,
    const float* __restrict__ g2, const float* __restrict__ be2,
    const float* __restrict__ m2, const float* __restrict__ v2,
    const float* __restrict__ w3, const float* __restrict__ b3,
    const float* __restrict__ g3, const float* __restrict__ be3,
    const float* __restrict__ m3, const float* __restrict__ v3,
    unsigned short* __restrict__ w1bf, unsigned short* __restrict__ w2bf,
    unsigned short* __restrict__ w3bf,
    float* __restrict__ sc1, float* __restrict__ sh1,
    float* __restrict__ sc2, float* __restrict__ sh2,
    float* __restrict__ sc3, float* __restrict__ sh3) {
  int t = blockIdx.x * 256 + threadIdx.x;
  if (t < M1 * K1) w1bf[t] = f2bf(w1[t]);
  if (t < M2 * M1) w2bf[t] = f2bf(w2[t]);
  if (t < 16 * 32) w3bf[t] = (t < M3 * 32) ? f2bf(w3[t]) : (unsigned short)0;
  if (t < M1) {
    float s = g1[t] * rsqrtf(v1[t] + EPSV);
    sc1[t] = s; sh1[t] = b1[t] * s + be1[t] - m1[t] * s;
  }
  if (t < M2) {
    float s = g2[t] * rsqrtf(v2[t] + EPSV);
    sc2[t] = s; sh2[t] = b2[t] * s + be2[t] - m2[t] * s;
  }
  if (t < M3) {
    float s = g3[t] * rsqrtf(v3[t] + EPSV);
    sc3[t] = s; sh3[t] = b3[t] * s + be3[t] - m3[t] * s;
  }
}

// ---------------------------------------------------------------
// Ego: feats[b,0] -> bf16 (two elements per thread, dword stores)
// ---------------------------------------------------------------
__global__ __launch_bounds__(256) void ego_kernel(
    const float* __restrict__ feats, unsigned short* __restrict__ ego) {
  long long i2 = (long long)blockIdx.x * 256 + threadIdx.x;  // over B*C*HW/2
  long long i = i2 * 2;
  int b = (int)(i / ((long long)CC * HWSZ));
  long long rem = i - (long long)b * CC * HWSZ;
  const float* s = feats + (long long)b * NN * CC * HWSZ + rem;
  *(unsigned int*)(ego + i) = pack_bf16(s[0], s[1]);
}

// ---------------------------------------------------------------
// Warp: bilinear affine resample -> neigh (bf16), 2 pixels/thread
// grid (HW/512, B*N)
// ---------------------------------------------------------------
__global__ __launch_bounds__(256) void warp_kernel(
    const float* __restrict__ feats, const float* __restrict__ t4,
    unsigned short* __restrict__ neigh) {
  const int bn = blockIdx.y;
  const int pix0 = (blockIdx.x * 256 + threadIdx.x) * 2;  // even, same row pair

  const float* T = t4 + bn * 16;
  const float th00 = T[0];
  const float th01 = T[1] * ((float)HH / (float)WW);
  const float th02 = T[3] * (2.0f / (W0F * DRATIO));
  const float th10 = T[4] * ((float)WW / (float)HH);
  const float th11 = T[5];
  const float th12 = T[7] * (2.0f / (H0F * DRATIO));

  int i00[2], i01[2], i10[2], i11[2];
  float m00[2], m01[2], m10[2], m11[2];
#pragma unroll
  for (int q = 0; q < 2; ++q) {
    const int pix = pix0 + q;
    const int x = pix % WW, y = pix / WW;
    const float xs = (2.0f * x + 1.0f) / WW - 1.0f;
    const float ys = (2.0f * y + 1.0f) / HH - 1.0f;
    const float gx = th00 * xs + th01 * ys + th02;
    const float gy = th10 * xs + th11 * ys + th12;
    const float px = ((gx + 1.0f) * WW - 1.0f) * 0.5f;
    const float py = ((gy + 1.0f) * HH - 1.0f) * 0.5f;
    const float fx0 = floorf(px), fy0 = floorf(py);
    const int x0 = (int)fx0, y0 = (int)fy0, x1 = x0 + 1, y1 = y0 + 1;
    const float wx1 = px - fx0, wy1 = py - fy0;
    const float wx0 = 1.0f - wx1, wy0 = 1.0f - wy1;
    const bool vx0 = (x0 >= 0) & (x0 < WW), vx1 = (x1 >= 0) & (x1 < WW);
    const bool vy0 = (y0 >= 0) & (y0 < HH), vy1 = (y1 >= 0) & (y1 < HH);
    const int cx0 = min(max(x0, 0), WW - 1), cx1 = min(max(x1, 0), WW - 1);
    const int cy0 = min(max(y0, 0), HH - 1), cy1 = min(max(y1, 0), HH - 1);
    m00[q] = (vx0 && vy0) ? wy0 * wx0 : 0.0f;
    m01[q] = (vx1 && vy0) ? wy0 * wx1 : 0.0f;
    m10[q] = (vx0 && vy1) ? wy1 * wx0 : 0.0f;
    m11[q] = (vx1 && vy1) ? wy1 * wx1 : 0.0f;
    i00[q] = cy0 * WW + cx0; i01[q] = cy0 * WW + cx1;
    i10[q] = cy1 * WW + cx0; i11[q] = cy1 * WW + cx1;
  }

  const size_t base = (size_t)bn * CC * HWSZ;
  const float* src = feats + base;
  unsigned short* dst = neigh + base + pix0;
#pragma unroll 4
  for (int c = 0; c < CC; ++c) {
    const float* s = src + (size_t)c * HWSZ;
    const float v0 = m00[0] * s[i00[0]] + m01[0] * s[i01[0]] +
                     m10[0] * s[i10[0]] + m11[0] * s[i11[0]];
    const float v1 = m00[1] * s[i00[1]] + m01[1] * s[i01[1]] +
                     m10[1] * s[i10[1]] + m11[1] * s[i11[1]];
    *(unsigned int*)(dst + (size_t)c * HWSZ) = pack_bf16(v0, v1);
  }
}

// ---------------------------------------------------------------
// Fused conv chain 768->128->32->8->1 with WMMA bf16, per (b,n,128px)
// grid (HW/128, B*N), block 256 (8 waves)
// Double-buffered LDS staging: one barrier per k-chunk; next chunk's
// global loads held in registers across the current chunk's WMMAs.
// ---------------------------------------------------------------
__global__ __launch_bounds__(256) void conv_kernel(
    const unsigned short* __restrict__ neigh, const unsigned short* __restrict__ ego,
    const unsigned short* __restrict__ w1bf, const unsigned short* __restrict__ w2bf,
    const unsigned short* __restrict__ w3bf,
    const float* __restrict__ sc1, const float* __restrict__ sh1,
    const float* __restrict__ sc2, const float* __restrict__ sh2,
    const float* __restrict__ sc3, const float* __restrict__ sh3,
    const float* __restrict__ w4, const float* __restrict__ b4,
    float* __restrict__ scores) {
  // pair-interleaved bf16 tiles: dword row = k-pair, 128 pixel columns
  __shared__ __align__(16) unsigned int ldsXs[2 * 16 * PIXB];  // 16KB stage x2
  __shared__ __align__(16) unsigned int ldsX1[64 * PIXB];      // 32KB
  __shared__ __align__(16) unsigned int ldsX2[16 * PIXB];      // 8KB

  const int tid = threadIdx.x;
  const int lane = tid & 31;
  const int wave = tid >> 5;
  const int half = lane >> 4;     // 0: lanes 0-15, 1: lanes 16-31
  const int lm = lane & 15;

  const int bn = blockIdx.y;
  const int pixbase = blockIdx.x * PIXB;

  const unsigned short* xn = neigh + (size_t)bn * CC * HWSZ;
  const unsigned short* xe = ego + (size_t)(bn / NN) * CC * HWSZ;

  // ---------------- GEMM1: 128 x 128 = W1(128x768) * X(768x128) --------------
  v8f acc[NT];
#pragma unroll
  for (int t = 0; t < NT; ++t) acc[t] = zero8();

  const int mrow1 = 16 * wave + lm;          // A-frag row (out channel)
  const int kbh = half * 8;                  // A-frag K offset per lane half
  const int k2s = tid >> 4;                  // stage: channel pair 0..15
  const int pgs = (tid & 15) << 3;           // stage: 8-pixel group 0,8,..,120
  const size_t rowoff = (size_t)(2 * k2s) * HWSZ + pixbase + pgs;

  uint4 sa, sb;  // register-staged next chunk (two channel rows x 8 px)
  {
    const unsigned short* r = xn + rowoff;
    sa = *(const uint4*)r;
    sb = *(const uint4*)(r + HWSZ);
  }

#pragma unroll 1
  for (int chunk = 0; chunk < 24; ++chunk) {
    unsigned int* stg = ldsXs + (chunk & 1) * (16 * PIXB);
    {   // pair-interleave the two staged channel rows, store 32B to LDS
      uint4 d0, d1;
      d0.x = ilv_lo(sa.x, sb.x); d0.y = ilv_hi(sa.x, sb.x);
      d0.z = ilv_lo(sa.y, sb.y); d0.w = ilv_hi(sa.y, sb.y);
      d1.x = ilv_lo(sa.z, sb.z); d1.y = ilv_hi(sa.z, sb.z);
      d1.z = ilv_lo(sa.w, sb.w); d1.w = ilv_hi(sa.w, sb.w);
      *(uint4*)(&stg[k2s * PIXB + pgs]) = d0;
      *(uint4*)(&stg[k2s * PIXB + pgs + 4]) = d1;
    }
    if (chunk < 23) {   // issue next chunk's global loads before the barrier
      const int cn = chunk + 1;
      const unsigned short* r = ((cn < 12) ? (xn + (size_t)cn * 32 * HWSZ)
                                           : (xe + (size_t)(cn - 12) * 32 * HWSZ)) + rowoff;
      if (chunk < 22) {  // distance-2 prefetch (global_prefetch_b8)
        const int cp = chunk + 2;
        const unsigned short* rp = ((cp < 12) ? (xn + (size_t)cp * 32 * HWSZ)
                                              : (xe + (size_t)(cp - 12) * 32 * HWSZ)) + rowoff;
        __builtin_prefetch(rp, 0, 3);
        __builtin_prefetch(rp + HWSZ, 0, 3);
      }
      sa = *(const uint4*)r;
      sb = *(const uint4*)(r + HWSZ);
    }
    __syncthreads();

    FragBF afr;
    {   // A fragment: W1 row mrow1, K chunk (contiguous k = 16*half + s)
      const unsigned int* wrow =
          (const unsigned int*)(w1bf + (size_t)mrow1 * K1 + chunk * 32);
#pragma unroll
      for (int v = 0; v < 8; ++v) {
        int kin = 2 * kbh + 2 * v;
        afr.u[v] = wrow[kin >> 1];
      }
    }
#pragma unroll
    for (int t = 0; t < NT; ++t) {
      FragBF bfr;
      const int p = t * 16 + lm;
#pragma unroll
      for (int r = 0; r < 8; ++r) bfr.u[r] = stg[(kbh + r) * PIXB + p];
      acc[t] = __builtin_amdgcn_wmma_f32_16x16x32_bf16(
          false, afr.v, false, bfr.v, (short)0, acc[t], false, false);
    }
  }

  // BN1 + ReLU -> bf16 x1 tile in LDS
  {
    const int mc = 16 * wave;
#pragma unroll
    for (int r = 0; r < 8; r += 2) {
      const int ch0 = mc + r + 8 * half;     // even
      const float2 sv = *(const float2*)(sc1 + ch0);
      const float2 hv = *(const float2*)(sh1 + ch0);
#pragma unroll
      for (int t = 0; t < NT; ++t) {
        const float y0 = fmaxf(acc[t][r] * sv.x + hv.x, 0.0f);
        const float y1 = fmaxf(acc[t][r + 1] * sv.y + hv.y, 0.0f);
        ldsX1[(ch0 >> 1) * PIXB + t * 16 + lm] = pack_bf16(y0, y1);
      }
    }
  }
  __syncthreads();

  // ------------- GEMM2: 32 x 128 = W2(32x128) * x1(128x128) -------------
  // 16 (m,n) tile-tasks over 8 waves: each wave does one m-tile, two n-tiles
  const int mt = wave & 1;
  const int ntA = wave >> 1;           // 0..3
  const int pA = ntA * 16 + lm;
  const int pB = pA + 64;              // n-tiles 4..7
  v8f acc2[2] = {zero8(), zero8()};
#pragma unroll
  for (int c2 = 0; c2 < 4; ++c2) {
    FragBF afr;
    const unsigned int* wrow =
        (const unsigned int*)(w2bf + (size_t)(mt * 16 + lm) * M1 + c2 * 32);
#pragma unroll
    for (int v = 0; v < 8; ++v) {
      int kin = 2 * kbh + 2 * v;
      afr.u[v] = wrow[kin >> 1];
    }
    FragBF bfrA, bfrB;
#pragma unroll
    for (int r = 0; r < 8; ++r) {
      bfrA.u[r] = ldsX1[(c2 * 16 + kbh + r) * PIXB + pA];
      bfrB.u[r] = ldsX1[(c2 * 16 + kbh + r) * PIXB + pB];
    }
    acc2[0] = __builtin_amdgcn_wmma_f32_16x16x32_bf16(
        false, afr.v, false, bfrA.v, (short)0, acc2[0], false, false);
    acc2[1] = __builtin_amdgcn_wmma_f32_16x16x32_bf16(
        false, afr.v, false, bfrB.v, (short)0, acc2[1], false, false);
  }
  {   // BN2 + ReLU -> bf16 x2 tile in LDS
#pragma unroll
    for (int r = 0; r < 8; r += 2) {
      const int ch0 = mt * 16 + r + 8 * half;
      const float2 sv = *(const float2*)(sc2 + ch0);
      const float2 hv = *(const float2*)(sh2 + ch0);
#pragma unroll
      for (int q = 0; q < 2; ++q) {
        const float y0 = fmaxf(acc2[q][r] * sv.x + hv.x, 0.0f);
        const float y1 = fmaxf(acc2[q][r + 1] * sv.y + hv.y, 0.0f);
        ldsX2[(ch0 >> 1) * PIXB + (q ? pB : pA)] = pack_bf16(y0, y1);
      }
    }
  }
  __syncthreads();

  // --------- GEMM3 (M padded 8->16) + conv4 + score: 8 n-tiles / 8 waves -----
  {
    const int p = wave * 16 + lm;
    FragBF afr, bfr;
    const unsigned int* wrow = (const unsigned int*)(w3bf + (size_t)lm * 32);
#pragma unroll
    for (int v = 0; v < 8; ++v) {
      int kin = 2 * kbh + 2 * v;
      afr.u[v] = wrow[kin >> 1];
    }
#pragma unroll
    for (int r = 0; r < 8; ++r) bfr.u[r] = ldsX2[(kbh + r) * PIXB + p];
    v8f acc3 = __builtin_amdgcn_wmma_f32_16x16x32_bf16(
        false, afr.v, false, bfr.v, (short)0, zero8(), false, false);
    // lanes 0-15 hold real channels 0..7 in regs 0..7 for pixel p
    if (lane < 16) {
      float s = b4[0];
#pragma unroll
      for (int r = 0; r < 8; ++r) {
        const float y = fmaxf(acc3[r] * sc3[r] + sh3[r], 0.0f);
        s += w4[r] * y;
      }
      scores[(size_t)bn * HWSZ + pixbase + p] = fmaxf(s, 0.0f);
    }
  }
}

// ---------------------------------------------------------------
// Softmax over N + weighted fusion. 2 pixels per thread.
// grid (B*HW/512)
// ---------------------------------------------------------------
__global__ __launch_bounds__(256) void fuse_kernel(
    const unsigned short* __restrict__ neigh, const float* __restrict__ scores,
    float* __restrict__ out) {
  const long long gid = (long long)blockIdx.x * 256 + threadIdx.x;
  const long long pix2 = gid * 2;
  const int b = (int)(pix2 / HWSZ);
  const int pix = (int)(pix2 - (long long)b * HWSZ);

  float s0[NN], s1[NN];
  float mx0 = -1e30f, mx1 = -1e30f;
#pragma unroll
  for (int n = 0; n < NN; ++n) {
    const float* sp = scores + (size_t)(b * NN + n) * HWSZ + pix;
    s0[n] = sp[0]; s1[n] = sp[1];
    mx0 = fmaxf(mx0, s0[n]); mx1 = fmaxf(mx1, s1[n]);
  }
  float sum0 = 0.0f, sum1 = 0.0f;
#pragma unroll
  for (int n = 0; n < NN; ++n) {
    s0[n] = __expf(s0[n] - mx0); sum0 += s0[n];
    s1[n] = __expf(s1[n] - mx1); sum1 += s1[n];
  }
  const float inv0 = 1.0f / sum0, inv1 = 1.0f / sum1;
#pragma unroll
  for (int n = 0; n < NN; ++n) { s0[n] *= inv0; s1[n] *= inv1; }

  for (int c = 0; c < CC; ++c) {
    float a0 = 0.0f, a1 = 0.0f;
#pragma unroll
    for (int n = 0; n < NN; ++n) {
      const unsigned u = *(const unsigned*)(
          neigh + ((size_t)(b * NN + n) * CC + c) * HWSZ + pix);
      a0 += s0[n] * bflo(u);
      a1 += s1[n] * bfhi(u);
    }
    float2 o; o.x = a0; o.y = a1;
    *(float2*)(out + ((size_t)b * CC + c) * HWSZ + pix) = o;
  }
}

// ---------------------------------------------------------------
extern "C" void kernel_launch(void* const* d_in, const int* in_sizes, int n_in,
                              void* d_out, int out_size, void* d_ws, size_t ws_size,
                              hipStream_t stream) {
  const float* feats = (const float*)d_in[0];
  const float* t4    = (const float*)d_in[1];
  const float* w1 = (const float*)d_in[2];  const float* b1 = (const float*)d_in[3];
  const float* g1 = (const float*)d_in[4];  const float* be1 = (const float*)d_in[5];
  const float* m1 = (const float*)d_in[6];  const float* v1 = (const float*)d_in[7];
  const float* w2 = (const float*)d_in[8];  const float* b2 = (const float*)d_in[9];
  const float* g2 = (const float*)d_in[10]; const float* be2 = (const float*)d_in[11];
  const float* m2 = (const float*)d_in[12]; const float* v2 = (const float*)d_in[13];
  const float* w3 = (const float*)d_in[14]; const float* b3 = (const float*)d_in[15];
  const float* g3 = (const float*)d_in[16]; const float* be3 = (const float*)d_in[17];
  const float* m3 = (const float*)d_in[18]; const float* v3 = (const float*)d_in[19];
  const float* w4 = (const float*)d_in[20]; const float* b4 = (const float*)d_in[21];

  char* ws = (char*)d_ws;
  size_t off = 0;
  unsigned short* neigh = (unsigned short*)(ws + off); off += (size_t)BB * NN * CC * HWSZ * 2;
  unsigned short* ego   = (unsigned short*)(ws + off); off += (size_t)BB * CC * HWSZ * 2;
  float* scores         = (float*)(ws + off);          off += (size_t)BB * NN * HWSZ * 4;
  unsigned short* w1bf  = (unsigned short*)(ws + off); off += (size_t)M1 * K1 * 2;
  unsigned short* w2bf  = (unsigned short*)(ws + off); off += (size_t)M2 * M1 * 2;
  unsigned short* w3bf  = (unsigned short*)(ws + off); off += (size_t)16 * 32 * 2;
  float* sc1 = (float*)(ws + off); off += M1 * 4;
  float* sh1 = (float*)(ws + off); off += M1 * 4;
  float* sc2 = (float*)(ws + off); off += M2 * 4;
  float* sh2 = (float*)(ws + off); off += M2 * 4;
  float* sc3 = (float*)(ws + off); off += M3 * 4;
  float* sh3 = (float*)(ws + off); off += M3 * 4;

  prep_kernel<<<(M1 * K1) / 256, 256, 0, stream>>>(
      w1, b1, g1, be1, m1, v1, w2, b2, g2, be2, m2, v2, w3, b3, g3, be3, m3, v3,
      w1bf, w2bf, w3bf, sc1, sh1, sc2, sh2, sc3, sh3);

  ego_kernel<<<(BB * CC * HWSZ / 2) / 256, 256, 0, stream>>>(feats, ego);

  warp_kernel<<<dim3(HWSZ / 512, BB * NN), 256, 0, stream>>>(feats, t4, neigh);

  conv_kernel<<<dim3(HWSZ / PIXB, BB * NN), 256, 0, stream>>>(
      neigh, ego, w1bf, w2bf, w3bf, sc1, sh1, sc2, sh2, sc3, sh3, w4, b4, scores);

  fuse_kernel<<<(BB * HWSZ / 2) / 256, 256, 0, stream>>>(neigh, scores, (float*)d_out);
}
